// HistEncoder_89867895701737
// MI455X (gfx1250) — matrix operand order repaired and checked by reading, compiled, not run on hardware
//
#include <hip/hip_runtime.h>
#include <hip/hip_bf16.h>
#include <math.h>

typedef __attribute__((ext_vector_type(16))) _Float16 v16h;
typedef __attribute__((ext_vector_type(8)))  _Float16 v8h;
typedef __attribute__((ext_vector_type(8)))  float    v8f;

#define T_LEN 16
#define B_AG  128
#define NTOTN 2048
#define NSEQ  2176           // B_AG + NTOTN sequences, length T_LEN
#define DMO   256
#define FFO   1024
#define NHEAD 8
#define DHEAD 32
#define NSLOT 39
#define NFEAT 4
#define TOK   (T_LEN*NSEQ)   // 34816 tokens

// ---------------------------------------------------------------------------
// WMMA GEMM: C[M,N] = act(A[M,K] * BT[N,K]^T + bias). A,BT f16 row-major.
// One wave (32 threads) computes a 16x16 tile. Writes f32 C and/or f16 C16.
// ACT: 0 = none, 1 = relu
// ---------------------------------------------------------------------------
template<int ACT>
__global__ void gemm16(const _Float16* __restrict__ A, const _Float16* __restrict__ BT,
                       const float* __restrict__ bias, float* __restrict__ C,
                       _Float16* __restrict__ C16, int M, int K, int N) {
  const int tile_m = blockIdx.x;
  const int tile_n = blockIdx.y;
  const int lane = threadIdx.x;
  const int half = lane >> 4;
  const int l16  = lane & 15;
  const int mrow = tile_m * 16 + l16;   // A row for this lane (both halves)
  const int nrow = tile_n * 16 + l16;   // BT row (= B column)
  const _Float16* arow = A  + (size_t)mrow * K;
  const _Float16* brow = BT + (size_t)nrow * K;
  v8f acc = {};
  for (int k0 = 0; k0 < K; k0 += 32) {
    union { v16h v; v8h h[2]; } a, b;
    // A 16x32 f16 layout: lane half 0 -> K = k0+0..7 & k0+16..23 ; half 1 -> +8
    a.h[0] = *(const v8h*)(arow + k0 + 8 * half);
    a.h[1] = *(const v8h*)(arow + k0 + 16 + 8 * half);
    // B 32x16 f16 layout: lane half 0 -> K = k0+0..7 & k0+16..23 ; half 1 -> +8
    b.h[0] = *(const v8h*)(brow + k0 + 8 * half);
    b.h[1] = *(const v8h*)(brow + k0 + 16 + 8 * half);
    __builtin_prefetch(arow + k0 + 64, 0, 0);
    acc = __builtin_amdgcn_wmma_f32_16x16x32_f16(false, a.v, false, b.v,
                                                 (short)0, acc, false, false);
  }
  const int col = tile_n * 16 + l16;
  const float bv = bias ? bias[col] : 0.0f;
#pragma unroll
  for (int r = 0; r < 8; ++r) {
    const int row = tile_m * 16 + r + 8 * half;   // C/D layout: M = r + 8*half
    float v = acc[r] + bv;
    if (ACT == 1) v = v > 0.f ? v : 0.f;
    const size_t idx = (size_t)row * N + col;
    if (C)   C[idx]   = v;
    if (C16) C16[idx] = (_Float16)v;
  }
}

// ---------------------------------------------------------------------------
// Input embedding: emb = feats(4) @ in_w + in_b ; x = leaky_relu(emb, 0.1)
// token = t*NSEQ + n ; n<B_AG -> src[n,t,:], else nbrs[n-B_AG,t,:]
// ---------------------------------------------------------------------------
__global__ void embed_kernel(const float* __restrict__ src, const float* __restrict__ nbrs,
                             const float* __restrict__ in_w, const float* __restrict__ in_b,
                             float* __restrict__ X, _Float16* __restrict__ X16,
                             float* __restrict__ src_emb) {
  const int tok = blockIdx.x, d = threadIdx.x;
  const int t = tok / NSEQ, n = tok % NSEQ;
  const float* f = (n < B_AG) ? (src  + ((size_t)n * T_LEN + t) * NFEAT)
                              : (nbrs + ((size_t)(n - B_AG) * T_LEN + t) * NFEAT);
  float e = in_b[d];
#pragma unroll
  for (int c = 0; c < NFEAT; ++c) e += f[c] * in_w[c * DMO + d];
  if (n < B_AG) src_emb[((size_t)t * B_AG + n) * DMO + d] = e;
  const float x = e > 0.f ? e : 0.1f * e;
  const size_t i = (size_t)tok * DMO + d;
  X[i] = x; X16[i] = (_Float16)x;
}

// qk16 = f16(x + sine positional encoding); pos recomputed on the fly
__global__ void make_qk16(const float* __restrict__ X, const float* __restrict__ src,
                          const float* __restrict__ nbrs, _Float16* __restrict__ QK16) {
  const int tok = blockIdx.x, d = threadIdx.x;
  const int t = tok / NSEQ, n = tok % NSEQ;
  const float* f = (n < B_AG) ? (src  + ((size_t)n * T_LEN + t) * NFEAT)
                              : (nbrs + ((size_t)(n - B_AG) * T_LEN + t) * NFEAT);
  const int dd = (d < 128) ? d : d - 128;
  const float coord = (d < 128) ? f[1] : f[0];          // (pos_y, pos_x)
  const float freq = __powf(10000.0f, (float)(dd & ~1) * (1.0f / 128.0f));
  const float p = coord * 6.283185307179586f / freq;
  const float pe = (dd & 1) ? __cosf(p) : __sinf(p);
  QK16[(size_t)tok * DMO + d] = (_Float16)(X[(size_t)tok * DMO + d] + pe);
}

// ---------------------------------------------------------------------------
// Encoder self-attention: per sequence n, per head (one wave), T=16, dh=32.
// ---------------------------------------------------------------------------
__global__ void enc_attn(const float* __restrict__ Q, const float* __restrict__ Kt,
                         const float* __restrict__ V, _Float16* __restrict__ O16) {
  const int n = blockIdx.x;
  const int h = threadIdx.x >> 5;
  const int lane = threadIdx.x & 31;
  __shared__ float att[NHEAD][T_LEN][T_LEN];
  const int dcol = h * DHEAD + lane;
  float kc[T_LEN], vc[T_LEN];
#pragma unroll
  for (int s = 0; s < T_LEN; ++s) {
    const size_t r = ((size_t)s * NSEQ + n) * DMO + dcol;
    kc[s] = Kt[r]; vc[s] = V[r];
  }
  const float scale = 0.17677669529663689f;  // 1/sqrt(32)
  for (int tq = 0; tq < T_LEN; ++tq) {
    const float qd = Q[((size_t)tq * NSEQ + n) * DMO + dcol];
#pragma unroll
    for (int ts = 0; ts < T_LEN; ++ts) {
      float p = qd * kc[ts];
      for (int m = 16; m > 0; m >>= 1) p += __shfl_xor(p, m, 32);
      if (lane == ts) att[h][tq][ts] = p * scale;
    }
  }
  for (int tq = 0; tq < T_LEN; ++tq) {
    float s = (lane < T_LEN) ? att[h][tq][lane] : -1e30f;
    float mx = s;
    for (int m = 8; m > 0; m >>= 1) mx = fmaxf(mx, __shfl_xor(mx, m, 32));
    float e = (lane < T_LEN) ? __expf(s - mx) : 0.f;
    float z = e;
    for (int m = 8; m > 0; m >>= 1) z += __shfl_xor(z, m, 32);
    if (lane < T_LEN) att[h][tq][lane] = e / z;
  }
  for (int tq = 0; tq < T_LEN; ++tq) {
    float od = 0.f;
#pragma unroll
    for (int ts = 0; ts < T_LEN; ++ts) od += att[h][tq][ts] * vc[ts];
    O16[((size_t)tq * NSEQ + n) * DMO + dcol] = (_Float16)od;
  }
}

// ---------------------------------------------------------------------------
// LayerNorm(X + R) over last dim (256). One row per block of 256 threads.
// ---------------------------------------------------------------------------
__global__ void ln_residual(const float* __restrict__ X, const float* __restrict__ R,
                            const float* __restrict__ g, const float* __restrict__ b,
                            float* __restrict__ Y, _Float16* __restrict__ Y16) {
  const int row = blockIdx.x, d = threadIdx.x;
  const size_t idx = (size_t)row * DMO + d;
  const float v = X[idx] + R[idx];
  __shared__ float red[DMO];
  red[d] = v; __syncthreads();
  for (int s = 128; s > 0; s >>= 1) { if (d < s) red[d] += red[d + s]; __syncthreads(); }
  const float mu = red[0] * (1.0f / DMO); __syncthreads();
  red[d] = (v - mu) * (v - mu); __syncthreads();
  for (int s = 128; s > 0; s >>= 1) { if (d < s) red[d] += red[d + s]; __syncthreads(); }
  const float var = red[0] * (1.0f / DMO);
  const float y = (v - mu) * rsqrtf(var + 1e-5f) * g[d] + b[d];
  if (Y)   Y[idx]   = y;
  if (Y16) Y16[idx] = (_Float16)y;
}

// masked_scatter index maps (feature-uniform masks -> prefix-count gather)
__global__ void scan_masks(const int* __restrict__ tmask, const int* __restrict__ smask,
                           int* __restrict__ idx_t, int* __restrict__ idx_s) {
  if (threadIdx.x | blockIdx.x) return;
  int c = 0;
  for (int i = 0; i < B_AG * NSLOT; ++i) {
    if (tmask[(size_t)i * NFEAT] != 0) { idx_t[i] = (c < NTOTN) ? c : NTOTN - 1; ++c; }
    else idx_t[i] = -1;
  }
  c = 0;
  for (int i = 0; i < B_AG * NSLOT; ++i) {
    if (smask[(size_t)i * DMO] != 0) { idx_s[i] = (c < NTOTN) ? c : NTOTN - 1; ++c; }
    else idx_s[i] = -1;
  }
}

__global__ void build_srcT(const float* __restrict__ src, float* __restrict__ srcT) {
  const int i = blockIdx.x * blockDim.x + threadIdx.x;  // T*B*4
  if (i >= T_LEN * B_AG * NFEAT) return;
  const int c = i & 3, r = i >> 2, b = r % B_AG, t = r / B_AG;
  srcT[i] = src[((size_t)b * T_LEN + t) * NFEAT + c];
}

__global__ void build_tgrid(const float* __restrict__ nbrs, const int* __restrict__ idx_t,
                            float* __restrict__ tgrid) {
  const size_t i = (size_t)blockIdx.x * blockDim.x + threadIdx.x;
  if (i >= (size_t)T_LEN * B_AG * NSLOT * NFEAT) return;
  const int c = i & 3; const size_t r = i >> 2;
  const int s = r % NSLOT; const size_t r2 = r / NSLOT;
  const int b = r2 % B_AG, t = r2 / B_AG;
  const int g = idx_t[b * NSLOT + s];
  tgrid[i] = (g >= 0) ? nbrs[((size_t)g * T_LEN + t) * NFEAT + c] : 0.0f;
}

__global__ void build_sgrid16(const _Float16* __restrict__ X16, const int* __restrict__ idx_s,
                              _Float16* __restrict__ sg) {
  const size_t i = (size_t)blockIdx.x * blockDim.x + threadIdx.x;
  if (i >= (size_t)T_LEN * B_AG * NSLOT * DMO) return;
  const int d = i & 255; const size_t r = i >> 8;
  const int s = r % NSLOT; const size_t r2 = r / NSLOT;
  const int b = r2 % B_AG, t = r2 / B_AG;
  const int g = idx_s[b * NSLOT + s];
  sg[i] = (g >= 0) ? X16[((size_t)t * NSEQ + B_AG + g) * DMO + d] : (_Float16)0.0f;
}

__global__ void slice_hist(const _Float16* __restrict__ X16, _Float16* __restrict__ H16) {
  const size_t i = (size_t)blockIdx.x * blockDim.x + threadIdx.x;  // T*B*DM
  if (i >= (size_t)T_LEN * B_AG * DMO) return;
  const int d = i & 255; const size_t r = i >> 8;
  const int n = r % B_AG, t = r / B_AG;
  H16[i] = X16[((size_t)t * NSEQ + n) * DMO + d];
}

// K=4 projection: Y[m,:] = X[m,0:4] @ W[4,256] + b
__global__ void proj4(const float* __restrict__ X, const float* __restrict__ W,
                      const float* __restrict__ b, float* __restrict__ Y) {
  const int m = blockIdx.x, d = threadIdx.x;
  const float* x = X + (size_t)m * NFEAT;
  float a = b[d];
#pragma unroll
  for (int c = 0; c < NFEAT; ++c) a += x[c] * W[c * DMO + d];
  Y[(size_t)m * DMO + d] = a;
}

// Grid attention over 39 slots: one wave per (t,b,head)
__global__ void grid_attn(const float* __restrict__ Q, const float* __restrict__ Kt,
                          const float* __restrict__ V, float* __restrict__ O,
                          _Float16* __restrict__ O16, float scale) {
  const int tb = blockIdx.x;
  const int h = threadIdx.x >> 5;
  const int lane = threadIdx.x & 31;
  __shared__ float sc[NHEAD][NSLOT];
  const int dcol = h * DHEAD + lane;
  const float qd = Q[(size_t)tb * DMO + dcol];
  for (int s = 0; s < NSLOT; ++s) {
    float p = qd * Kt[((size_t)tb * NSLOT + s) * DMO + dcol];
    for (int m = 16; m > 0; m >>= 1) p += __shfl_xor(p, m, 32);
    if (lane == 0) sc[h][s] = p * scale;
  }
  float mx = -1e30f;
  for (int s = 0; s < NSLOT; ++s) mx = fmaxf(mx, sc[h][s]);
  float z = 0.f;
  for (int s = 0; s < NSLOT; ++s) z += __expf(sc[h][s] - mx);
  float od = 0.f;
  for (int s = 0; s < NSLOT; ++s)
    od += __expf(sc[h][s] - mx) * V[((size_t)tb * NSLOT + s) * DMO + dcol];
  od /= z;
  const size_t oi = (size_t)tb * DMO + dcol;
  if (O)   O[oi]   = od;
  if (O16) O16[oi] = (_Float16)od;
}

// agg = lrelu(t_res + s_out); out[b,t,:] = concat(agg, hist_enc)
__global__ void finalize(const float* __restrict__ t_res, const float* __restrict__ s_out,
                         const float* __restrict__ X, float* __restrict__ out) {
  const size_t i = (size_t)blockIdx.x * blockDim.x + threadIdx.x;  // T*B*DM
  if (i >= (size_t)T_LEN * B_AG * DMO) return;
  const int d = i & 255; const size_t r = i >> 8;
  const int b = r % B_AG, t = r / B_AG;
  float a = t_res[i] + s_out[i];
  a = a > 0.f ? a : 0.1f * a;
  const float hist = X[((size_t)t * NSEQ + b) * DMO + d];
  const size_t ob = ((size_t)b * T_LEN + t) * 2 * DMO;
  out[ob + d] = a;
  out[ob + DMO + d] = hist;
}

// f32 W[K,N] -> f16 WT[N,K]
__global__ void convtrans(const float* __restrict__ W, _Float16* __restrict__ WT,
                          int K, int N) {
  const size_t i = (size_t)blockIdx.x * blockDim.x + threadIdx.x;
  if (i >= (size_t)K * N) return;
  const int n = (int)(i % N), k = (int)(i / N);
  WT[(size_t)n * K + k] = (_Float16)W[i];
}

// ---------------------------------------------------------------------------
extern "C" void kernel_launch(void* const* d_in, const int* in_sizes, int n_in,
                              void* d_out, int out_size, void* d_ws, size_t ws_size,
                              hipStream_t stream) {
  const float* src      = (const float*)d_in[0];
  const float* nbrs     = (const float*)d_in[1];
  const int*   smask    = (const int*)d_in[2];
  const int*   tmask    = (const int*)d_in[3];
  const float* in_w     = (const float*)d_in[4];
  const float* in_b     = (const float*)d_in[5];
  const float* enc_wqkv = (const float*)d_in[6];
  const float* enc_bqkv = (const float*)d_in[7];
  const float* enc_wo   = (const float*)d_in[8];
  const float* enc_bo   = (const float*)d_in[9];
  const float* ln1g     = (const float*)d_in[10];
  const float* ln1b     = (const float*)d_in[11];
  const float* enc_w1   = (const float*)d_in[12];
  const float* enc_b1   = (const float*)d_in[13];
  const float* enc_w2   = (const float*)d_in[14];
  const float* enc_b2   = (const float*)d_in[15];
  const float* ln2g     = (const float*)d_in[16];
  const float* ln2b     = (const float*)d_in[17];
  const float* tq_w = (const float*)d_in[18]; const float* tq_b = (const float*)d_in[19];
  const float* tk_w = (const float*)d_in[20]; const float* tk_b = (const float*)d_in[21];
  const float* tv_w = (const float*)d_in[22]; const float* tv_b = (const float*)d_in[23];
  const float* res_w = (const float*)d_in[24]; const float* res_b = (const float*)d_in[25];
  const float* res_ln_g = (const float*)d_in[26]; const float* res_ln_b = (const float*)d_in[27];
  const float* qf_w = (const float*)d_in[28]; const float* qf_b = (const float*)d_in[29];
  const float* kf_w = (const float*)d_in[30]; const float* kf_b = (const float*)d_in[31];
  const float* vf_w = (const float*)d_in[32]; const float* vf_b = (const float*)d_in[33];

  char* ws = (char*)d_ws;
  size_t off = 0;
  auto alloc = [&](size_t bytes) -> void* {
    void* p = ws + off;
    off += (bytes + 255) & ~(size_t)255;
    return p;
  };

  const size_t TD  = (size_t)TOK * DMO;          // 34816*256
  const size_t GBD = (size_t)T_LEN * B_AG * DMO; // 2048*256
  const size_t SKV = (size_t)T_LEN * B_AG * NSLOT * DMO; // 79872*256

  // f16 transposed weights
  _Float16* Wqkv16 = (_Float16*)alloc((size_t)3 * 3 * 65536 * 2);
  _Float16* Wo16   = (_Float16*)alloc((size_t)3 * 65536 * 2);
  _Float16* W1T16  = (_Float16*)alloc((size_t)3 * 262144 * 2);
  _Float16* W2T16  = (_Float16*)alloc((size_t)3 * 262144 * 2);
  _Float16* Res16  = (_Float16*)alloc(65536 * 2);
  _Float16* Qf16   = (_Float16*)alloc(65536 * 2);
  _Float16* Kf16   = (_Float16*)alloc(65536 * 2);
  _Float16* Vf16   = (_Float16*)alloc(65536 * 2);
  // activations
  float*     X    = (float*)alloc(TD * 4);
  _Float16*  X16  = (_Float16*)alloc(TD * 2);
  _Float16*  QK16 = (_Float16*)alloc(TD * 2);
  float*     Qb   = (float*)alloc(TD * 4);
  float*     Kb   = (float*)alloc(TD * 4);
  float*     Vb   = (float*)alloc(TD * 4);
  _Float16*  O16  = (_Float16*)alloc(TD * 2);
  float*     TMP  = (float*)alloc(TD * 4);
  _Float16*  H16  = (_Float16*)alloc((size_t)TOK * FFO * 2);  // reused as s_grid16
  float*     SRCE = (float*)alloc(GBD * 4);
  int*       IDXT = (int*)alloc(B_AG * NSLOT * 4);
  int*       IDXS = (int*)alloc(B_AG * NSLOT * 4);
  float*     SRCT = (float*)alloc((size_t)T_LEN * B_AG * NFEAT * 4);
  float*     TGRD = (float*)alloc((size_t)T_LEN * B_AG * NSLOT * NFEAT * 4);
  float*     GQ   = (float*)alloc(GBD * 4);
  float*     GK   = (float*)alloc(SKV * 4);
  float*     GV   = (float*)alloc(SKV * 4);
  float*     GO   = (float*)alloc(GBD * 4);
  _Float16*  GO16 = (_Float16*)alloc(GBD * 2);
  float*     TPJ  = (float*)alloc(GBD * 4);
  float*     TRES = (float*)alloc(GBD * 4);
  _Float16*  HIS16 = (_Float16*)alloc(GBD * 2);
  _Float16*  SG16 = H16;  // reuse (FFN hidden dead by social phase)
  if (off > ws_size) return;  // workspace too small; nothing safe to do

  // index maps for masked_scatter
  scan_masks<<<1, 32, 0, stream>>>(tmask, smask, IDXT, IDXS);

  // weight conversion (f32 [K,N] -> f16 [N,K])
  for (int l = 0; l < 3; ++l) {
    for (int j = 0; j < 3; ++j)
      convtrans<<<256, 256, 0, stream>>>(enc_wqkv + (size_t)(l * 3 + j) * 65536,
                                         Wqkv16 + (size_t)(l * 3 + j) * 65536, DMO, DMO);
    convtrans<<<256, 256, 0, stream>>>(enc_wo + (size_t)l * 65536, Wo16 + (size_t)l * 65536, DMO, DMO);
    convtrans<<<1024, 256, 0, stream>>>(enc_w1 + (size_t)l * 262144, W1T16 + (size_t)l * 262144, DMO, FFO);
    convtrans<<<1024, 256, 0, stream>>>(enc_w2 + (size_t)l * 262144, W2T16 + (size_t)l * 262144, FFO, DMO);
  }
  convtrans<<<256, 256, 0, stream>>>(res_w, Res16, DMO, DMO);
  convtrans<<<256, 256, 0, stream>>>(qf_w, Qf16, DMO, DMO);
  convtrans<<<256, 256, 0, stream>>>(kf_w, Kf16, DMO, DMO);
  convtrans<<<256, 256, 0, stream>>>(vf_w, Vf16, DMO, DMO);

  // embedding (+ leaky relu) and raw src_emb
  embed_kernel<<<TOK, DMO, 0, stream>>>(src, nbrs, in_w, in_b, X, X16, SRCE);

  // shared encoder over concatenated [src | nbrs] sequences
  const dim3 gBlk(32);
  for (int l = 0; l < 3; ++l) {
    make_qk16<<<TOK, DMO, 0, stream>>>(X, src, nbrs, QK16);
    gemm16<0><<<dim3(TOK / 16, DMO / 16), gBlk, 0, stream>>>(
        QK16, Wqkv16 + (size_t)(l * 3 + 0) * 65536, enc_bqkv + (size_t)(l * 3 + 0) * DMO,
        Qb, nullptr, TOK, DMO, DMO);
    gemm16<0><<<dim3(TOK / 16, DMO / 16), gBlk, 0, stream>>>(
        QK16, Wqkv16 + (size_t)(l * 3 + 1) * 65536, enc_bqkv + (size_t)(l * 3 + 1) * DMO,
        Kb, nullptr, TOK, DMO, DMO);
    gemm16<0><<<dim3(TOK / 16, DMO / 16), gBlk, 0, stream>>>(
        X16, Wqkv16 + (size_t)(l * 3 + 2) * 65536, enc_bqkv + (size_t)(l * 3 + 2) * DMO,
        Vb, nullptr, TOK, DMO, DMO);
    enc_attn<<<NSEQ, 256, 0, stream>>>(Qb, Kb, Vb, O16);
    gemm16<0><<<dim3(TOK / 16, DMO / 16), gBlk, 0, stream>>>(
        O16, Wo16 + (size_t)l * 65536, enc_bo + (size_t)l * DMO, TMP, nullptr, TOK, DMO, DMO);
    ln_residual<<<TOK, DMO, 0, stream>>>(X, TMP, ln1g + (size_t)l * DMO, ln1b + (size_t)l * DMO, X, X16);
    gemm16<1><<<dim3(TOK / 16, FFO / 16), gBlk, 0, stream>>>(
        X16, W1T16 + (size_t)l * 262144, enc_b1 + (size_t)l * FFO, nullptr, H16, TOK, DMO, FFO);
    gemm16<0><<<dim3(TOK / 16, DMO / 16), gBlk, 0, stream>>>(
        H16, W2T16 + (size_t)l * 262144, enc_b2 + (size_t)l * DMO, TMP, nullptr, TOK, FFO, DMO);
    ln_residual<<<TOK, DMO, 0, stream>>>(X, TMP, ln2g + (size_t)l * DMO, ln2b + (size_t)l * DMO, X, X16);
  }

  // ---- temporal grid attention ----
  build_srcT<<<(T_LEN * B_AG * NFEAT + 255) / 256, 256, 0, stream>>>(src, SRCT);
  build_tgrid<<<(T_LEN * B_AG * NSLOT * NFEAT + 255) / 256, 256, 0, stream>>>(nbrs, IDXT, TGRD);
  proj4<<<T_LEN * B_AG, DMO, 0, stream>>>(SRCT, tq_w, tq_b, GQ);
  proj4<<<T_LEN * B_AG * NSLOT, DMO, 0, stream>>>(TGRD, tk_w, tk_b, GK);
  proj4<<<T_LEN * B_AG * NSLOT, DMO, 0, stream>>>(TGRD, tv_w, tv_b, GV);
  grid_attn<<<T_LEN * B_AG, 256, 0, stream>>>(GQ, GK, GV, nullptr, GO16,
                                              0.17677669529663689f);  // 1/sqrt(32)
  gemm16<0><<<dim3((T_LEN * B_AG) / 16, DMO / 16), gBlk, 0, stream>>>(
      GO16, Res16, res_b, TPJ, nullptr, T_LEN * B_AG, DMO, DMO);
  ln_residual<<<T_LEN * B_AG, DMO, 0, stream>>>(SRCE, TPJ, res_ln_g, res_ln_b, TRES, nullptr);

  // ---- social grid attention ----
  slice_hist<<<(int)(GBD / 256), 256, 0, stream>>>(X16, HIS16);
  build_sgrid16<<<(int)((SKV + 255) / 256), 256, 0, stream>>>(X16, IDXS, SG16);
  gemm16<0><<<dim3((T_LEN * B_AG) / 16, DMO / 16), gBlk, 0, stream>>>(
      HIS16, Qf16, qf_b, GQ, nullptr, T_LEN * B_AG, DMO, DMO);
  gemm16<0><<<dim3((T_LEN * B_AG * NSLOT) / 16, DMO / 16), gBlk, 0, stream>>>(
      SG16, Kf16, kf_b, GK, nullptr, T_LEN * B_AG * NSLOT, DMO, DMO);
  gemm16<0><<<dim3((T_LEN * B_AG * NSLOT) / 16, DMO / 16), gBlk, 0, stream>>>(
      SG16, Vf16, vf_b, GV, nullptr, T_LEN * B_AG * NSLOT, DMO, DMO);
  grid_attn<<<T_LEN * B_AG, 256, 0, stream>>>(GQ, GK, GV, GO, nullptr,
                                              0.0625f);  // 1/sqrt(256)

  // ---- output: [B, T, 2*DM] ----
  finalize<<<(int)(GBD / 256), 256, 0, stream>>>(TRES, GO, X, (float*)d_out);
}
